// MILModel_33097017983355
// MI455X (gfx1250) — matrix operand order repaired and checked by reading, not compiled, not run on hardware
//
#include <hip/hip_runtime.h>
#include <hip/hip_bf16.h>

typedef __attribute__((ext_vector_type(16))) _Float16 v16h;
typedef __attribute__((ext_vector_type(8)))  _Float16 v8h;
typedef __attribute__((ext_vector_type(8)))  float    v8f;
typedef __attribute__((ext_vector_type(4)))  int      v4i;

#define B_SZ   8192
#define DIN    2048
#define C_SZ   1000
#define CPAD   1024
#define P_SZ   256

union V16U { v16h v; v8h h[2]; };

__device__ __forceinline__ v8f wmma_f16(v16h a, v16h b, v8f c) {
  // D = A(16x32 f16) * B(32x16 f16) + C(16x16 f32)
  return __builtin_amdgcn_wmma_f32_16x16x32_f16(
      /*neg_a=*/false, a, /*neg_b=*/false, b,
      /*c_mod=*/(short)0, c, /*reuse_a=*/false, /*reuse_b=*/false);
}

// 16-byte global -> LDS async DMA (ASYNCcnt-tracked, no VGPR staging)
__device__ __forceinline__ void async_b128(const _Float16* g, _Float16* l) {
  __builtin_amdgcn_global_load_async_to_lds_b128(
      (__attribute__((address_space(1))) v4i*)g,
      (__attribute__((address_space(3))) v4i*)l,
      /*offset=*/0, /*cpol=*/0);
}

// ---------------- f32 -> f16 convert with row/col zero padding ----------------
__global__ __launch_bounds__(256) void cvt_pad_kernel(
    const float* __restrict__ src, _Float16* __restrict__ dst,
    int srcRows, int dstRows, int srcCols, int dstCols) {
  size_t n = (size_t)dstRows * (size_t)dstCols;
  size_t stride = (size_t)gridDim.x * blockDim.x;
  for (size_t i = (size_t)blockIdx.x * blockDim.x + threadIdx.x; i < n; i += stride) {
    int r = (int)(i / (size_t)dstCols);
    int c = (int)(i % (size_t)dstCols);
    float val = (r < srcRows && c < srcCols) ? src[(size_t)r * srcCols + c] : 0.0f;
    dst[i] = (_Float16)val;
  }
}

// ---------------- GEMM1: preds = x @ Wm^T + bm ----------------
// block = 256 threads (8 waves), tile 128(M) x 128(N), BK = 32, double-buffered
// async-DMA LDS staging. wave (mw,nw): mw 0..3 -> 32-row strip, nw 0..1 -> 64-col strip.
#define BM 128
#define BN 128
#define BK 32
#define KTILES (DIN / BK)

__global__ __launch_bounds__(256) void gemm1_kernel(
    const _Float16* __restrict__ x16, const _Float16* __restrict__ w16,
    const float* __restrict__ bm, float* __restrict__ preds,
    _Float16* __restrict__ preds16) {
  __shared__ _Float16 As[2][BM][BK] __attribute__((aligned(128)));
  __shared__ _Float16 Bs[2][BN][BK] __attribute__((aligned(128)));

  const int tid  = threadIdx.x;
  const int lane = tid & 31;
  const int wave = tid >> 5;
  const int mw   = wave & 3;   // 0..3
  const int nw   = wave >> 2;  // 0..1
  const int lm   = lane & 15;  // row (A) / col (B,C)
  const int lh   = lane >> 4;  // K-half selector
  const int rowBase = blockIdx.y * BM;
  const int colBase = blockIdx.x * BN;

  v8f acc[2][4];
#pragma unroll
  for (int i = 0; i < 2; ++i)
#pragma unroll
    for (int j = 0; j < 4; ++j) acc[i][j] = (v8f)0.0f;

  // issue stage 0: each thread DMAs 2 A-chunks + 2 B-chunks (16B each)
#pragma unroll
  for (int ch = tid; ch < 512; ch += 256) {
    int r = ch >> 2, c8 = (ch & 3) << 3;
    async_b128(&x16[(size_t)(rowBase + r) * DIN + c8], &As[0][r][c8]);
    async_b128(&w16[(size_t)(colBase + r) * DIN + c8], &Bs[0][r][c8]);
  }

  for (int kt = 0; kt < KTILES; ++kt) {
    const int buf = kt & 1;
    if (kt + 1 < KTILES) {
      const int k0n = (kt + 1) * BK;
#pragma unroll
      for (int ch = tid; ch < 512; ch += 256) {
        int r = ch >> 2, c8 = (ch & 3) << 3;
        async_b128(&x16[(size_t)(rowBase + r) * DIN + k0n + c8], &As[buf ^ 1][r][c8]);
        async_b128(&w16[(size_t)(colBase + r) * DIN + k0n + c8], &Bs[buf ^ 1][r][c8]);
      }
      // async loads complete in order: <=4 outstanding == current tile landed
      __builtin_amdgcn_s_wait_asynccnt(4);
    } else {
      __builtin_amdgcn_s_wait_asynccnt(0);
    }
    __syncthreads();  // every wave has waited -> full tile visible in LDS

    v16h afrag[2];
#pragma unroll
    for (int i = 0; i < 2; ++i) {
      int m = mw * 32 + i * 16 + lm;
      V16U u;
      u.h[0] = *(const v8h*)&As[buf][m][lh * 16];      // K 0-7  / 16-23
      u.h[1] = *(const v8h*)&As[buf][m][lh * 16 + 8];  // K 8-15 / 24-31
      afrag[i] = u.v;
    }
    v16h bfrag[4];
#pragma unroll
    for (int j = 0; j < 4; ++j) {
      int n = nw * 64 + j * 16 + lm;
      bfrag[j] = *(const v16h*)&Bs[buf][n][lh * 16];   // K 0-15 / 16-31 contiguous
    }
#pragma unroll
    for (int i = 0; i < 2; ++i)
#pragma unroll
      for (int j = 0; j < 4; ++j)
        acc[i][j] = wmma_f16(afrag[i], bfrag[j], acc[i][j]);
    __syncthreads();  // all waves done reading buf before next DMA overwrites it
  }

  // epilogue: add bias, write f32 preds (B x 1000) and f16 preds (B x 1024, padded)
#pragma unroll
  for (int j = 0; j < 4; ++j) {
    int cg = colBase + nw * 64 + j * 16 + lm;
    float bias = (cg < C_SZ) ? bm[cg] : 0.0f;
#pragma unroll
    for (int i = 0; i < 2; ++i) {
#pragma unroll
      for (int e = 0; e < 8; ++e) {
        int rg = rowBase + mw * 32 + i * 16 + e + 8 * lh;
        float val = acc[i][j][e] + bias;
        if (cg < C_SZ) {
          preds[(size_t)rg * C_SZ + cg] = val;
          preds16[(size_t)rg * CPAD + cg] = (_Float16)val;
        } else {
          preds16[(size_t)rg * CPAD + cg] = (_Float16)0.0f;
        }
      }
    }
  }
}

// ---------------- GEMM2: gated = tanh(preds@u^T) * sigmoid(preds@v^T) ----------------
// block = 256 threads (8 waves). Tile 32(M) x 64(N). Wave: 16x16 against BOTH u and v.
__global__ __launch_bounds__(256) void gemm2_kernel(
    const _Float16* __restrict__ p16, const _Float16* __restrict__ u16,
    const _Float16* __restrict__ v16p, float* __restrict__ gated) {
  const int lane = threadIdx.x & 31;
  const int wave = threadIdx.x >> 5;
  const int mw   = wave >> 2;  // 0..1
  const int nw   = wave & 3;   // 0..3
  const int lm   = lane & 15;
  const int lh   = lane >> 4;
  const int rBase = blockIdx.y * 32 + mw * 16;
  const int cBase = blockIdx.x * 64 + nw * 16;

  v8f accU = (v8f)0.0f, accV = (v8f)0.0f;
  const size_t arow = (size_t)(rBase + lm) * CPAD;
  const size_t brow = (size_t)(cBase + lm) * CPAD;

  for (int k0 = 0; k0 < CPAD; k0 += 32) {
    V16U ua;
    ua.h[0] = *(const v8h*)&p16[arow + k0 + lh * 16];
    ua.h[1] = *(const v8h*)&p16[arow + k0 + lh * 16 + 8];
    v16h bu = *(const v16h*)&u16[brow + k0 + lh * 16];
    v16h bv = *(const v16h*)&v16p[brow + k0 + lh * 16];
    accU = wmma_f16(ua.v, bu, accU);
    accV = wmma_f16(ua.v, bv, accV);
  }
#pragma unroll
  for (int e = 0; e < 8; ++e) {
    int rg = rBase + e + 8 * lh;
    int cg = cBase + lm;
    float t = tanhf(accU[e]);
    float s = 1.0f / (1.0f + expf(-accV[e]));
    gated[(size_t)rg * P_SZ + cg] = t * s;
  }
}

// ---------------- scores[b] = gated[b,:] . w ; one wave per row ----------------
__global__ __launch_bounds__(256) void scores_kernel(
    const float* __restrict__ gated, const float* __restrict__ w,
    float* __restrict__ scores) {
  int row  = blockIdx.x * 8 + (threadIdx.x >> 5);
  int lane = threadIdx.x & 31;
  float s = 0.0f;
#pragma unroll
  for (int p = lane; p < P_SZ; p += 32) s += gated[(size_t)row * P_SZ + p] * w[p];
#pragma unroll
  for (int off = 16; off > 0; off >>= 1) s += __shfl_xor(s, off, 32);
  if (lane == 0) scores[row] = s;
}

// ---------------- rowsum[b] = sum_c preds[b,c] ; one wave per row ----------------
__global__ __launch_bounds__(256) void rowsum_kernel(
    const float* __restrict__ preds, float* __restrict__ rowsum) {
  int row  = blockIdx.x * 8 + (threadIdx.x >> 5);
  int lane = threadIdx.x & 31;
  float s = 0.0f;
  for (int c = lane; c < C_SZ; c += 32) s += preds[(size_t)row * C_SZ + c];
#pragma unroll
  for (int off = 16; off > 0; off >>= 1) s += __shfl_xor(s, off, 32);
  if (lane == 0) rowsum[row] = s;
}

// ---------------- softmax over scores + weighted rowsum -> total_pred ----------------
__global__ __launch_bounds__(1024) void finalize_kernel(
    const float* __restrict__ scores, const float* __restrict__ rowsum,
    float* __restrict__ out_total) {
  __shared__ float smax[1024];
  __shared__ float ssum[1024];
  __shared__ float sdot[1024];
  int tid = threadIdx.x;
  float m = -3.4e38f;
  for (int i = tid; i < B_SZ; i += 1024) m = fmaxf(m, scores[i]);
  smax[tid] = m;
  __syncthreads();
  for (int s = 512; s > 0; s >>= 1) {
    if (tid < s) smax[tid] = fmaxf(smax[tid], smax[tid + s]);
    __syncthreads();
  }
  float M = smax[0];
  float se = 0.0f, dot = 0.0f;
  for (int i = tid; i < B_SZ; i += 1024) {
    float e = expf(scores[i] - M);
    se += e;
    dot += e * rowsum[i];
  }
  ssum[tid] = se; sdot[tid] = dot;
  __syncthreads();
  for (int s = 512; s > 0; s >>= 1) {
    if (tid < s) { ssum[tid] += ssum[tid + s]; sdot[tid] += sdot[tid + s]; }
    __syncthreads();
  }
  if (tid == 0) out_total[0] = sdot[0] / ssum[0];
}

extern "C" void kernel_launch(void* const* d_in, const int* in_sizes, int n_in,
                              void* d_out, int out_size, void* d_ws, size_t ws_size,
                              hipStream_t stream) {
  const float* x  = (const float*)d_in[0];  // (8192, 2048)
  const float* Wm = (const float*)d_in[1];  // (1000, 2048)
  const float* bm = (const float*)d_in[2];  // (1000,)
  const float* u  = (const float*)d_in[3];  // (256, 1000)
  const float* v  = (const float*)d_in[4];  // (256, 1000)
  const float* w  = (const float*)d_in[5];  // (256, 1)

  float* preds = (float*)d_out;                       // (8192, 1000)
  float* total = preds + (size_t)B_SZ * C_SZ;         // scalar

  // workspace layout (bytes)
  char* ws = (char*)d_ws;
  _Float16* x16 = (_Float16*)ws;                             // 8192x2048
  _Float16* w16 = (_Float16*)(ws + 33554432);                // 1024x2048 (row-padded)
  _Float16* u16 = (_Float16*)(ws + 33554432 + 4194304);      // 256x1024 (col-padded)
  _Float16* v16 = (_Float16*)(ws + 33554432 + 4194304 + 524288);
  _Float16* p16 = (_Float16*)(ws + 33554432 + 4194304 + 2 * 524288);          // 8192x1024
  float* gated  = (float*)(ws + 33554432 + 4194304 + 2 * 524288 + 16777216);  // 8192x256
  float* scores = (float*)((char*)gated + 8388608);          // 8192
  float* rowsum = (float*)((char*)scores + 32768);           // 8192

  // 1) convert + pad to f16
  cvt_pad_kernel<<<4096, 256, 0, stream>>>(x,  x16, B_SZ, B_SZ, DIN, DIN);
  cvt_pad_kernel<<<2048, 256, 0, stream>>>(Wm, w16, C_SZ, CPAD, DIN, DIN);
  cvt_pad_kernel<<<512,  256, 0, stream>>>(u,  u16, P_SZ, P_SZ, C_SZ, CPAD);
  cvt_pad_kernel<<<512,  256, 0, stream>>>(v,  v16, P_SZ, P_SZ, C_SZ, CPAD);

  // 2) preds = x @ Wm^T + bm    (WMMA f16, f32 accumulate, async-DMA double buffer)
  gemm1_kernel<<<dim3(CPAD / BN, B_SZ / BM), 256, 0, stream>>>(x16, w16, bm, preds, p16);

  // 3) rowsum of preds
  rowsum_kernel<<<B_SZ / 8, 256, 0, stream>>>(preds, rowsum);

  // 4) gated attention GEMMs (WMMA f16) with fused tanh*sigmoid
  gemm2_kernel<<<dim3(P_SZ / 64, B_SZ / 32), 256, 0, stream>>>(p16, u16, v16, gated);

  // 5) scores, softmax, weighted sum
  scores_kernel<<<B_SZ / 8, 256, 0, stream>>>(gated, w, scores);
  finalize_kernel<<<1, 1024, 0, stream>>>(scores, rowsum, total);
}
